// GATDiagnosis_28432683499569
// MI455X (gfx1250) — compile-verified, matrix operand order checked by reading, not run on hardware
//
#include <hip/hip_runtime.h>
#include <stdint.h>
#include <stddef.h>

// ---------------------------------------------------------------------------
// GAT diagnosis pipeline for gfx1250 (MI455X, wave32, WMMA).
// GEMMs: bf16 WMMA 16x16x32 with f32 accumulation, operands pre-packed into
// ISA fragment order so the inner loop is b128 loads + v_wmma only.
// Edge phase: L2-resident gather + atomic segment softmax + atomic scatter.
// ---------------------------------------------------------------------------

typedef __attribute__((ext_vector_type(16))) __bf16 v16bf;
typedef __attribute__((ext_vector_type(8)))  float  v8f;

#define LDSTR 520  // 512 + 8-float pad: row stride of 520 dwords avoids LDS bank conflicts

// fp32 -> bf16 (round to nearest even), returned in low 16 bits
__device__ __forceinline__ unsigned f2bf(float f) {
  unsigned u = __float_as_uint(f);
  return (u + 0x7FFFu + ((u >> 16) & 1u)) >> 16;
}

// K offset of the even element held in dword v of a 16x16x32 bf16 WMMA A/B
// fragment for this lane (ISA 7.12.2: VGPR0..3 = K 0..7 / 8..15 by lane half,
// VGPR4..7 = +16).
__device__ __forceinline__ int kmap(int v, int lane) {
  return ((v & 4) << 2) | ((lane & 16) >> 1) | ((v & 3) << 1);
}

// Order-preserving f32 <-> u32 encoding so atomicMax(u32) implements float max.
// 0u is a below-everything sentinel (== encode(-inf) - eps region).
__device__ __forceinline__ unsigned fenc(float f) {
  unsigned u = __float_as_uint(f);
  return (u & 0x80000000u) ? ~u : (u | 0x80000000u);
}
__device__ __forceinline__ float fdec(unsigned u) {
  return __uint_as_float((u & 0x80000000u) ? (u & 0x7FFFFFFFu) : ~u);
}

// Repack a 16x512 f32 LDS tile into bf16 WMMA-A fragments (K=512 -> 16 ksteps).
// 256 threads: wave w handles ksteps 2w, 2w+1.
__device__ __forceinline__ void pack_tile_to_fragA(const float* tile,
                                                   unsigned* dst, int mt, int tid) {
  const int lane = tid & 31, w = tid >> 5;
  const int m = lane & 15;
#pragma unroll
  for (int q = 0; q < 2; ++q) {
    const int ks = w * 2 + q;
#pragma unroll
    for (int v = 0; v < 8; ++v) {
      const int k = (ks << 5) + kmap(v, lane);
      const unsigned lo = f2bf(tile[m * LDSTR + k]);
      const unsigned hi = f2bf(tile[m * LDSTR + k + 1]);
      dst[((size_t)(mt * 16 + ks) * 32 + lane) * 8 + v] = lo | (hi << 16);
    }
  }
}

// ---- weight / input pre-packing into fragment order -----------------------
// B source layout: row-major [K, N].  Pack index = ((nt*nks+ks)*32+lane)*8+v.
__global__ void k_pack_B(const float* __restrict__ W, unsigned* __restrict__ pack,
                         int K, int N) {
  const int id = blockIdx.x * 256 + threadIdx.x;
  const int nks = K >> 5;
  const int total = (N >> 4) * nks * 256;
  if (id >= total) return;
  const int v = id & 7, l = (id >> 3) & 31;
  const int t = id >> 8, ks = t % nks, nt = t / nks;
  const int n = (nt << 4) + (l & 15);
  const int k = (ks << 5) + kmap(v, l);
  pack[id] = f2bf(W[(size_t)k * N + n]) | (f2bf(W[(size_t)(k + 1) * N + n]) << 16);
}

// A source layout: row-major [M, K].
__global__ void k_pack_A(const float* __restrict__ X, unsigned* __restrict__ pack,
                         int M, int K) {
  const int id = blockIdx.x * 256 + threadIdx.x;
  const int nks = K >> 5;
  const int total = (M >> 4) * nks * 256;
  if (id >= total) return;
  const int v = id & 7, l = (id >> 3) & 31;
  const int t = id >> 8, ks = t % nks, mt = t / nks;
  const int m = (mt << 4) + (l & 15);
  const int k = (ks << 5) + kmap(v, l);
  pack[id] = f2bf(X[(size_t)m * K + k]) | (f2bf(X[(size_t)m * K + k + 1]) << 16);
}

// ---- kernel 1: x @ ft_W + b -> LayerNorm -> LeakyReLU -> bf16 A-fragments --
// block = 256 (8 waves), each block owns 16 rows x 512 cols; K = 128.
__global__ __launch_bounds__(256) void k_gemm_ft_ln(
    const unsigned* __restrict__ packX, const unsigned* __restrict__ packW,
    const float* __restrict__ ft_b, const float* __restrict__ ln_g,
    const float* __restrict__ ln_b, unsigned* __restrict__ packH1) {
  __shared__ float tile[16 * LDSTR];
  __shared__ float psum[16][16], psq[16][16], stats[16][2];
  const int tid = threadIdx.x, lane = tid & 31, w = tid >> 5;
  const int mt = blockIdx.x;
  const int nks = 4;  // K = 128
  v8f acc[4] = {};
  for (int ks = 0; ks < nks; ++ks) {
    v16bf a = *(const v16bf*)(packX + ((size_t)(mt * nks + ks) * 32 + lane) * 8);
#pragma unroll
    for (int j = 0; j < 4; ++j) {
      const int nt = w * 4 + j;
      v16bf b = *(const v16bf*)(packW + ((size_t)(nt * nks + ks) * 32 + lane) * 8);
      acc[j] = __builtin_amdgcn_wmma_f32_16x16x32_bf16(false, a, false, b,
                                                       (short)0, acc[j], false, false);
    }
  }
  // D layout: VGPR r holds m = r + 8*(lane>=16), col = tilebase + (lane&15)
  const int m0 = (lane >> 4) << 3, nc = lane & 15;
#pragma unroll
  for (int j = 0; j < 4; ++j) {
    const int n = (w * 4 + j) * 16 + nc;
    const float bias = ft_b[n];
#pragma unroll
    for (int r = 0; r < 8; ++r) tile[(m0 + r) * LDSTR + n] = acc[j][r] + bias;
  }
  __syncthreads();
  {  // LayerNorm row statistics
    const int row = tid >> 4, p = tid & 15;
    float s = 0.f, s2 = 0.f;
    for (int c = p * 32; c < p * 32 + 32; ++c) {
      const float xv = tile[row * LDSTR + c];
      s += xv; s2 += xv * xv;
    }
    psum[row][p] = s; psq[row][p] = s2;
  }
  __syncthreads();
  if (tid < 16) {
    float s = 0.f, s2 = 0.f;
    for (int p = 0; p < 16; ++p) { s += psum[tid][p]; s2 += psq[tid][p]; }
    const float mu = s * (1.f / 512.f);
    const float var = s2 * (1.f / 512.f) - mu * mu;
    stats[tid][0] = mu;
    stats[tid][1] = rsqrtf(var + 1e-5f);
  }
  __syncthreads();
  {  // normalize + affine + LeakyReLU(0.2)
    const int row = tid >> 4, p = tid & 15;
    const float mu = stats[row][0], rs = stats[row][1];
    for (int c = p * 32; c < p * 32 + 32; ++c) {
      float xv = (tile[row * LDSTR + c] - mu) * rs * ln_g[c] + ln_b[c];
      tile[row * LDSTR + c] = xv >= 0.f ? xv : 0.2f * xv;
    }
  }
  __syncthreads();
  pack_tile_to_fragA(tile, packH1, mt, tid);
}

// ---- kernel 2: h1 @ W1 -> hW1 (f32) + per-head attention scalars ----------
__global__ __launch_bounds__(256) void k_gemm1(
    const unsigned* __restrict__ packA, const unsigned* __restrict__ packW,
    const float* __restrict__ att_src, const float* __restrict__ att_dst,
    float* __restrict__ hW, float* __restrict__ a_src, float* __restrict__ a_dst) {
  __shared__ float tile[16 * LDSTR];
  const int tid = threadIdx.x, lane = tid & 31, w = tid >> 5;
  const int mt = blockIdx.x;
  const int nks = 16;  // K = 512
  v8f acc[4] = {};
  for (int ks = 0; ks < nks; ++ks) {
    v16bf a = *(const v16bf*)(packA + ((size_t)(mt * nks + ks) * 32 + lane) * 8);
#pragma unroll
    for (int j = 0; j < 4; ++j) {
      const int nt = w * 4 + j;
      v16bf b = *(const v16bf*)(packW + ((size_t)(nt * nks + ks) * 32 + lane) * 8);
      acc[j] = __builtin_amdgcn_wmma_f32_16x16x32_bf16(false, a, false, b,
                                                       (short)0, acc[j], false, false);
    }
  }
  const int m0 = (lane >> 4) << 3, nc = lane & 15;
#pragma unroll
  for (int j = 0; j < 4; ++j) {
    const int n = (w * 4 + j) * 16 + nc;
#pragma unroll
    for (int r = 0; r < 8; ++r) tile[(m0 + r) * LDSTR + n] = acc[j][r];
  }
  __syncthreads();
  for (int i = tid; i < 16 * 512; i += 256) {  // coalesced f32 store of hW1
    const int row = i >> 9, c = i & 511;
    hW[(size_t)(mt * 16 + row) * 512 + c] = tile[row * LDSTR + c];
  }
  if (tid < 128) {  // a_src/a_dst: [16 rows] x [8 heads], dot over C=64
    const int row = tid >> 3, h = tid & 7;
    float s1 = 0.f, s2 = 0.f;
    for (int c = 0; c < 64; ++c) {
      const float v = tile[row * LDSTR + (h << 6) + c];
      s1 += v * att_src[(h << 6) + c];
      s2 += v * att_dst[(h << 6) + c];
    }
    const int node = mt * 16 + row;
    a_src[node * 8 + h] = s1;
    a_dst[node * 8 + h] = s2;
  }
}

// ---- edge phase -----------------------------------------------------------
__global__ void k_edge_logit_max(const int* __restrict__ ei,
    const float* __restrict__ a_src, const float* __restrict__ a_dst,
    float* __restrict__ buf, unsigned* __restrict__ maxenc, int E) {
  const int idx = blockIdx.x * 256 + threadIdx.x;
  if (idx >= E * 8) return;
  const int e = idx >> 3, h = idx & 7;
  const int s = ei[e], d = ei[E + e];
  float L = a_src[s * 8 + h] + a_dst[d * 8 + h];
  L = L >= 0.f ? L : 0.2f * L;  // LeakyReLU(0.2)
  buf[idx] = L;
  atomicMax(&maxenc[d * 8 + h], fenc(L));
}

__global__ void k_edge_expsum(const int* __restrict__ ei,
    float* __restrict__ buf, const unsigned* __restrict__ maxenc,
    float* __restrict__ denom, int E) {
  const int idx = blockIdx.x * 256 + threadIdx.x;
  if (idx >= E * 8) return;
  const int e = idx >> 3, h = idx & 7;
  const int d = ei[E + e];
  const float ex = __expf(buf[idx] - fdec(maxenc[d * 8 + h]));
  buf[idx] = ex;
  atomicAdd(&denom[d * 8 + h], ex);
}

// layer-1 scatter: block per edge, 512 channels, L2-resident gather + atomics
__global__ __launch_bounds__(256) void k_scatter1(const int* __restrict__ ei,
    const float* __restrict__ hW, const float* __restrict__ expv,
    const float* __restrict__ denom, float* __restrict__ agg, int E) {
  __shared__ float alpha[8];
  const int e = blockIdx.x;
  const int s = ei[e], d = ei[E + e];
  if (threadIdx.x < 8)
    alpha[threadIdx.x] = expv[e * 8 + threadIdx.x] / denom[d * 8 + threadIdx.x];
  __syncthreads();
#pragma unroll
  for (int i = 0; i < 2; ++i) {
    const int c = threadIdx.x + i * 256;
    atomicAdd(&agg[(size_t)d * 512 + c], hW[(size_t)s * 512 + c] * alpha[c >> 6]);
  }
}

__global__ void k_scatter2(const int* __restrict__ ei,
    const float* __restrict__ hW, const float* __restrict__ expv,
    const float* __restrict__ denom, float* __restrict__ agg, int E) {
  const int idx = blockIdx.x * 256 + threadIdx.x;
  if (idx >= E * 16) return;
  const int e = idx >> 4, c = idx & 15, h = c >> 1;
  const int s = ei[e], d = ei[E + e];
  const float alpha = expv[e * 8 + h] / denom[d * 8 + h];
  atomicAdd(&agg[(size_t)d * 16 + c], hW[(size_t)s * 16 + c] * alpha);
}

// ---- bias + BatchNorm(eval) + ReLU -> bf16 A-fragments for GEMM2 ----------
__global__ __launch_bounds__(256) void k_bnprep(const float* __restrict__ agg,
    const float* __restrict__ b1, const float* __restrict__ bn_g,
    const float* __restrict__ bn_b, const float* __restrict__ bn_mean,
    const float* __restrict__ bn_var, unsigned* __restrict__ packH2) {
  __shared__ float tile[16 * LDSTR];
  const int mt = blockIdx.x, tid = threadIdx.x;
  for (int i = tid; i < 16 * 512; i += 256) {
    const int row = i >> 9, c = i & 511;
    float xv = agg[(size_t)(mt * 16 + row) * 512 + c] + b1[c];
    xv = (xv - bn_mean[c]) * rsqrtf(bn_var[c] + 1e-5f) * bn_g[c] + bn_b[c];
    tile[row * LDSTR + c] = xv > 0.f ? xv : 0.f;
  }
  __syncthreads();
  pack_tile_to_fragA(tile, packH2, mt, tid);
}

// ---- kernel 3: h2 @ W2 (512 -> 16), one wave per 16-row tile --------------
__global__ __launch_bounds__(32) void k_gemm2(
    const unsigned* __restrict__ packA, const unsigned* __restrict__ packW,
    const float* __restrict__ att_src, const float* __restrict__ att_dst,
    float* __restrict__ hW, float* __restrict__ a_src, float* __restrict__ a_dst) {
  __shared__ float t2[16][16];
  const int lane = threadIdx.x;
  const int mt = blockIdx.x;
  const int nks = 16;  // K = 512
  v8f acc = {};
  for (int ks = 0; ks < nks; ++ks) {
    v16bf a = *(const v16bf*)(packA + ((size_t)(mt * nks + ks) * 32 + lane) * 8);
    v16bf b = *(const v16bf*)(packW + ((size_t)ks * 32 + lane) * 8);
    acc = __builtin_amdgcn_wmma_f32_16x16x32_bf16(false, a, false, b,
                                                  (short)0, acc, false, false);
  }
  const int m0 = (lane >> 4) << 3, nc = lane & 15;
#pragma unroll
  for (int r = 0; r < 8; ++r) t2[m0 + r][nc] = acc[r];
  __syncthreads();
  for (int i = lane; i < 256; i += 32) {
    const int row = i >> 4, c = i & 15;
    hW[(size_t)(mt * 16 + row) * 16 + c] = t2[row][c];
  }
  if (lane < 16) {
    const int row = lane, node = mt * 16 + row;
    for (int h = 0; h < 8; ++h) {
      const float v0 = t2[row][2 * h], v1 = t2[row][2 * h + 1];
      a_src[node * 8 + h] = v0 * att_src[2 * h] + v1 * att_src[2 * h + 1];
      a_dst[node * 8 + h] = v0 * att_dst[2 * h] + v1 * att_dst[2 * h + 1];
    }
  }
}

// ---- final: mean over heads + b2 -----------------------------------------
__global__ void k_final(const float* __restrict__ agg, const float* __restrict__ b2,
                        float* __restrict__ out, int N) {
  const int t = blockIdx.x * 256 + threadIdx.x;
  if (t >= N * 2) return;
  const int n = t >> 1, c = t & 1;
  float s = 0.f;
#pragma unroll
  for (int h = 0; h < 8; ++h) s += agg[n * 16 + 2 * h + c];
  out[t] = s * 0.125f + b2[c];
}

__global__ void k_zero(unsigned* __restrict__ p, size_t n) {
  const size_t i = (size_t)blockIdx.x * 256 + threadIdx.x;
  if (i < n) p[i] = 0u;
}

// ---------------------------------------------------------------------------
extern "C" void kernel_launch(void* const* d_in, const int* in_sizes, int n_in,
                              void* d_out, int out_size, void* d_ws, size_t ws_size,
                              hipStream_t stream) {
  const float* x       = (const float*)d_in[0];
  const int*   ei      = (const int*)d_in[1];   // [2, E] (int32 per harness convention)
  const float* ft_W    = (const float*)d_in[2];
  const float* ft_b    = (const float*)d_in[3];
  const float* ln_g    = (const float*)d_in[4];
  const float* ln_b    = (const float*)d_in[5];
  const float* W1      = (const float*)d_in[6];
  const float* att_s1  = (const float*)d_in[7];
  const float* att_d1  = (const float*)d_in[8];
  const float* b1      = (const float*)d_in[9];
  const float* bn_g    = (const float*)d_in[10];
  const float* bn_b    = (const float*)d_in[11];
  const float* bn_mean = (const float*)d_in[12];
  const float* bn_var  = (const float*)d_in[13];
  const float* W2      = (const float*)d_in[14];
  const float* att_s2  = (const float*)d_in[15];
  const float* att_d2  = (const float*)d_in[16];
  const float* b2      = (const float*)d_in[17];

  const int N  = in_sizes[0] / 128;  // 10000 (multiple of 16)
  const int E  = in_sizes[1] / 2;    // 160000
  const int MT = N / 16;             // 625 row tiles

  // ---- workspace carve-up (~79 MB total) ----
  char* cur = (char*)d_ws;
  auto alloc = [&](size_t bytes) -> void* {
    void* p = (void*)cur;
    cur += (bytes + 255) & ~(size_t)255;
    return p;
  };
  unsigned* packWft = (unsigned*)alloc((size_t)32 * 4 * 256 * 4);   // ft_W frags
  unsigned* packW1  = (unsigned*)alloc((size_t)32 * 16 * 256 * 4);  // W1 frags
  unsigned* packW2  = (unsigned*)alloc((size_t)1 * 16 * 256 * 4);   // W2 frags
  unsigned* packX   = (unsigned*)alloc((size_t)MT * 4 * 256 * 4);   // x A-frags
  unsigned* packH1  = (unsigned*)alloc((size_t)MT * 16 * 256 * 4);  // LN(h) A-frags
  unsigned* packH2  = (unsigned*)alloc((size_t)MT * 16 * 256 * 4);  // BN(h) A-frags
  float* hW1 = (float*)alloc((size_t)N * 512 * 4);
  float* hW2 = (float*)alloc((size_t)N * 16 * 4);
  float* aS1 = (float*)alloc((size_t)N * 8 * 4);
  float* aD1 = (float*)alloc((size_t)N * 8 * 4);
  float* aS2 = (float*)alloc((size_t)N * 8 * 4);
  float* aD2 = (float*)alloc((size_t)N * 8 * 4);
  float* ex1 = (float*)alloc((size_t)E * 8 * 4);
  float* ex2 = (float*)alloc((size_t)E * 8 * 4);
  // contiguous zero-init region (accumulators + softmax state); cleared each call
  char* z0 = cur;
  float*    agg1 = (float*)alloc((size_t)N * 512 * 4);
  float*    agg2 = (float*)alloc((size_t)N * 16 * 4);
  unsigned* max1 = (unsigned*)alloc((size_t)N * 8 * 4);
  float*    den1 = (float*)alloc((size_t)N * 8 * 4);
  unsigned* max2 = (unsigned*)alloc((size_t)N * 8 * 4);
  float*    den2 = (float*)alloc((size_t)N * 8 * 4);
  const size_t zwords = (size_t)(cur - z0) / 4;

  k_zero<<<(unsigned)((zwords + 255) / 256), 256, 0, stream>>>((unsigned*)z0, zwords);

  // weight/input fragment packing (bf16)
  k_pack_B<<<(32 * 4 * 256 + 255) / 256, 256, 0, stream>>>(ft_W, packWft, 128, 512);
  k_pack_B<<<(32 * 16 * 256 + 255) / 256, 256, 0, stream>>>(W1, packW1, 512, 512);
  k_pack_B<<<(1 * 16 * 256 + 255) / 256, 256, 0, stream>>>(W2, packW2, 512, 16);
  k_pack_A<<<(MT * 4 * 256 + 255) / 256, 256, 0, stream>>>(x, packX, N, 128);

  // feature transform (GEMM + bias + LN + LeakyReLU) -> bf16 frags
  k_gemm_ft_ln<<<MT, 256, 0, stream>>>(packX, packWft, ft_b, ln_g, ln_b, packH1);

  // GAT layer 1
  k_gemm1<<<MT, 256, 0, stream>>>(packH1, packW1, att_s1, att_d1, hW1, aS1, aD1);
  const int EH = E * 8;
  k_edge_logit_max<<<(EH + 255) / 256, 256, 0, stream>>>(ei, aS1, aD1, ex1, max1, E);
  k_edge_expsum<<<(EH + 255) / 256, 256, 0, stream>>>(ei, ex1, max1, den1, E);
  k_scatter1<<<E, 256, 0, stream>>>(ei, hW1, ex1, den1, agg1, E);

  // bias + BN(eval) + ReLU -> bf16 frags
  k_bnprep<<<MT, 256, 0, stream>>>(agg1, b1, bn_g, bn_b, bn_mean, bn_var, packH2);

  // GAT layer 2
  k_gemm2<<<MT, 32, 0, stream>>>(packH2, packW2, att_s2, att_d2, hW2, aS2, aD2);
  k_edge_logit_max<<<(EH + 255) / 256, 256, 0, stream>>>(ei, aS2, aD2, ex2, max2, E);
  k_edge_expsum<<<(EH + 255) / 256, 256, 0, stream>>>(ei, ex2, max2, den2, E);
  k_scatter2<<<(E * 16 + 255) / 256, 256, 0, stream>>>(ei, hW2, ex2, den2, agg2, E);

  // head mean + b2 -> logits [N, 2]
  k_final<<<(N * 2 + 255) / 256, 256, 0, stream>>>(agg2, b2, (float*)d_out, N);
}
